// QRotationAttention_46952582480155
// MI455X (gfx1250) — matrix operand order repaired and checked by reading, not compiled, not run on hardware
//
#include <hip/hip_runtime.h>
#include <hip/hip_bf16.h>
#include <math.h>

// ---------------------------------------------------------------------------
// QRotationAttention for MI455X (gfx1250, wave32, WMMA).
//
// Pipeline (all f32 accumulate, f16 matrix operands via v_wmma_f32_16x16x32_f16):
//  1. k_quant_w      : per-row int8 fake-quant of Wq/Wk/Wv/Wp -> f16
//  2. k_quant_gamma  : global fake-quant of gamma -> f32
//  3. k_xpose        : x f32 [b][c][q][n] -> xh f16 [bq][n][c]
//                      (async global->LDS staging, ASYNCcnt fence, LDS transpose)
//  4. k_proj_qkv     : 3 GEMMs sharing the B (x) fragment; q,k -> [bq][c][n],
//                      v -> [bq][n][c] (transposed store = contiguous 16B/lane)
//  5. k_attn1        : S = Q K^T * 1/8 (K=1600), row softmax in 64KB LDS, P f16
//  6. k_attn2        : P @ V, stored transposed oT [bq][n][c]
//  7. k_projp        : Wp @ O + bp -> d_out (pre-BN, f32, final memory layout)
//  8. k_stats        : per-channel mean/var over (b,q,n)
//  9. k_bn           : in-place normalize with quantized gamma + beta
//
// Roofline: ~40 GFLOP total; all f16 intermediates (~136 MB) fit in the 192 MB
// global L2, so fragment loads re-hit L2; WMMA f16 w/ f32 accum keeps the
// matmuls on the matrix pipe. Workspace requirement: ~136 MB.
// ---------------------------------------------------------------------------

typedef __attribute__((ext_vector_type(16))) _Float16 v16h;
typedef __attribute__((ext_vector_type(8)))  _Float16 v8h;
typedef __attribute__((ext_vector_type(8)))  float    v8f;

#define B_   8
#define C_   256
#define Q_   4
#define N_   1600          /* H*W */
#define BQ_  32
#define EPSV 1e-5f

// ---------------- WMMA fragment helpers (CDNA5 16x16x32 f16 layouts) -------
// A (16x32 f16, MxK): lane L -> M = L&15; half = L>>4.
//   v[0..7]  hold K = half*8 + 0..7           (16B contiguous)
//   v[8..15] hold K = 16 + half*8 + 0..7      (16B contiguous, +32B)
// B (32x16 f16, KxN): lane L -> N = L&15; half = L>>4.
//   v[0..7]  hold K = half*8 + 0..7; v[8..15] hold K = 16 + half*8 + 0..7
// C/D (16x16 f32): lane L -> N = L&15; vgpr r -> M = 8*(L>>4) + r.

__device__ __forceinline__ v8f zero8() {
  v8f z;
#pragma unroll
  for (int i = 0; i < 8; ++i) z[i] = 0.0f;
  return z;
}

__device__ __forceinline__ v16h pack16(const _Float16* p0, const _Float16* p1) {
  v8h lo = *(const v8h*)p0;
  v8h hi = *(const v8h*)p1;
  v16h r;
#pragma unroll
  for (int i = 0; i < 8; ++i) { r[i] = lo[i]; r[i + 8] = hi[i]; }
  return r;
}

// A fragment from row-major [M x K] matrix, leading dim ld (f16 elements)
__device__ __forceinline__ v16h loadA(const _Float16* m, int ld, int row0, int k0, int lane) {
  const _Float16* p = m + (size_t)(row0 + (lane & 15)) * ld + k0 + ((lane >> 4) << 3);
  return pack16(p, p + 16);
}

// B fragment: B[k][n] sourced from row-major [N x K] storage (K contiguous)
__device__ __forceinline__ v16h loadB(const _Float16* m, int ld, int col0, int k0, int lane) {
  const _Float16* p = m + (size_t)(col0 + (lane & 15)) * ld + k0 + ((lane >> 4) << 3);
  return pack16(p, p + 16);
}

__device__ __forceinline__ v8f wmma_f16(v16h a, v16h b, v8f c) {
  return __builtin_amdgcn_wmma_f32_16x16x32_f16(false, a, false, b, (short)0, c, false, false);
}

// ---------------- 1) weight fake-quant (per output-channel row) ------------
__global__ void k_quant_w(const float* __restrict__ W, _Float16* __restrict__ Wh) {
  __shared__ float red[256];
  int row = blockIdx.x, t = threadIdx.x;
  float w = W[row * C_ + t];
  red[t] = fabsf(w);
  __syncthreads();
#pragma unroll
  for (int s = 128; s > 0; s >>= 1) {
    if (t < s) red[t] = fmaxf(red[t], red[t + s]);
    __syncthreads();
  }
  float sc = red[0] / 127.0f + 1e-8f;
  float q = fminf(fmaxf(rintf(w / sc), -127.0f), 127.0f) * sc;
  Wh[row * C_ + t] = (_Float16)q;
}

// ---------------- 2) gamma fake-quant (global) -----------------------------
__global__ void k_quant_gamma(const float* __restrict__ g, float* __restrict__ gq) {
  __shared__ float red[256];
  int t = threadIdx.x;
  float w = g[t];
  red[t] = fabsf(w);
  __syncthreads();
#pragma unroll
  for (int s = 128; s > 0; s >>= 1) {
    if (t < s) red[t] = fmaxf(red[t], red[t + s]);
    __syncthreads();
  }
  float sc = red[0] / 127.0f + 1e-8f;
  gq[t] = fminf(fmaxf(rintf(w / sc), -127.0f), 127.0f) * sc;
}

// ---------------- 3) x -> f16, transposed to [bq][n][c] --------------------
// Stages the 32x32 f32 tile into LDS with gfx1250 async global->LDS copies
// (no VGPR round-trip), fences with s_wait_asynccnt, then emits transposed f16.
__global__ void k_xpose(const float* __restrict__ x, _Float16* __restrict__ xh) {
  __shared__ float tile[32][33];
  int bq = blockIdx.z, b = bq >> 2, q = bq & 3;
  int n0 = blockIdx.x * 32, c0 = blockIdx.y * 32;
  int tx = threadIdx.x, ty = threadIdx.y;  // 32 x 8
#pragma unroll
  for (int i = 0; i < 32; i += 8) {
    int c = c0 + ty + i;
    const float* src = &x[(((size_t)b * C_ + c) * Q_ + q) * N_ + n0 + tx];
    unsigned lds = (unsigned)(uintptr_t)&tile[ty + i][tx];
    asm volatile("global_load_async_to_lds_b32 %0, %1, off"
                 :: "v"(lds), "v"(src)
                 : "memory");
  }
  asm volatile("s_wait_asynccnt 0x0" ::: "memory");
  __syncthreads();
#pragma unroll
  for (int i = 0; i < 32; i += 8) {
    int n = n0 + ty + i;
    xh[((size_t)bq * N_ + n) * C_ + c0 + tx] = (_Float16)tile[tx][ty + i];
  }
}

// ---------------- 4) fused q/k/v projection --------------------------------
// block = 256 thr (8 waves); block tile M=32 (o) x N=64 (n); 1 wave tile each.
__global__ void k_proj_qkv(const _Float16* __restrict__ xh,
                           const _Float16* __restrict__ Whq,
                           const _Float16* __restrict__ Whk,
                           const _Float16* __restrict__ Whv,
                           const float* __restrict__ biq,
                           const float* __restrict__ bik,
                           const float* __restrict__ biv,
                           _Float16* __restrict__ qh, _Float16* __restrict__ kh,
                           _Float16* __restrict__ vT) {
  int bq = blockIdx.z;
  int lane = threadIdx.x & 31, wid = threadIdx.x >> 5;
  int o0 = blockIdx.y * 32 + (wid & 1) * 16;
  int n0 = blockIdx.x * 64 + (wid >> 1) * 16;
  const _Float16* xb = xh + (size_t)bq * N_ * C_;
  v8f aq = zero8(), ak = zero8(), av = zero8();
  for (int k = 0; k < C_; k += 32) {
    v16h bf = loadB(xb, C_, n0, k, lane);   // x tile shared by the 3 GEMMs
    v16h a0 = loadA(Whq, C_, o0, k, lane);
    v16h a1 = loadA(Whk, C_, o0, k, lane);
    v16h a2 = loadA(Whv, C_, o0, k, lane);
    aq = wmma_f16(a0, bf, aq);
    ak = wmma_f16(a1, bf, ak);
    av = wmma_f16(a2, bf, av);
  }
  int n = n0 + (lane & 15);
  int mbase = o0 + ((lane >> 4) << 3);
#pragma unroll
  for (int r = 0; r < 8; ++r) {   // q,k: [c][n] layout (n-major for attn1)
    int o = mbase + r;
    qh[((size_t)bq * C_ + o) * N_ + n] = (_Float16)(aq[r] + biq[o]);
    kh[((size_t)bq * C_ + o) * N_ + n] = (_Float16)(ak[r] + bik[o]);
  }
  v8h vv;                          // v: [n][c] layout -> one 16B store/lane
#pragma unroll
  for (int r = 0; r < 8; ++r) vv[r] = (_Float16)(av[r] + biv[mbase + r]);
  *(v8h*)(vT + ((size_t)bq * N_ + n) * C_ + mbase) = vv;
}

// ---------------- 5) S = Q K^T * scale, softmax -> P -----------------------
// block = 256 thr; 64-row S block in 64KB dynamic LDS; grid (4, 32).
__global__ void k_attn1(const _Float16* __restrict__ qh,
                        const _Float16* __restrict__ kh,
                        _Float16* __restrict__ P) {
  extern __shared__ float S[];  // [64][256] fp32
  int bq = blockIdx.y;
  int c0 = blockIdx.x * 64;
  int lane = threadIdx.x & 31, wid = threadIdx.x >> 5;
  const _Float16* qb = qh + (size_t)bq * C_ * N_;
  const _Float16* kb = kh + (size_t)bq * C_ * N_;
  int mt = wid & 3;             // 4 row-tiles of 16
  int ntb = (wid >> 2) * 8;     // 2 wave-groups x 8 col-tiles
  v8f acc[8];
#pragma unroll
  for (int j = 0; j < 8; ++j) acc[j] = zero8();
  for (int k = 0; k < N_; k += 32) {
    v16h a = loadA(qb, N_, c0 + mt * 16, k, lane);
    __builtin_prefetch(qb + (size_t)(c0 + mt * 16 + (lane & 15)) * N_ + k + 128, 0, 0);
#pragma unroll
    for (int j = 0; j < 8; ++j) {
      v16h b = loadB(kb, N_, (ntb + j) * 16, k, lane);
      acc[j] = wmma_f16(a, b, acc[j]);
    }
  }
  const float scale = 0.125f;  // 1/sqrt(C/4) = 1/8
  int rbase = mt * 16 + ((lane >> 4) << 3);
#pragma unroll
  for (int j = 0; j < 8; ++j) {
    int col = (ntb + j) * 16 + (lane & 15);
#pragma unroll
    for (int r = 0; r < 8; ++r) S[(rbase + r) * C_ + col] = acc[j][r] * scale;
  }
  __syncthreads();
  if (threadIdx.x < 64) {       // one thread per S row: max / exp / sum / emit
    float* row = S + threadIdx.x * C_;
    float m = -1e30f;
    for (int d = 0; d < C_; ++d) m = fmaxf(m, row[d]);
    float s = 0.0f;
    for (int d = 0; d < C_; ++d) { float e = __expf(row[d] - m); row[d] = e; s += e; }
    float inv = 1.0f / s;
    _Float16* prow = P + ((size_t)bq * C_ + c0 + threadIdx.x) * C_;
    for (int d = 0; d < C_; ++d) prow[d] = (_Float16)(row[d] * inv);
  }
}

// ---------------- 6) O = P @ V, stored transposed [bq][n][c] ---------------
__global__ void k_attn2(const _Float16* __restrict__ P,
                        const _Float16* __restrict__ vT,
                        _Float16* __restrict__ oT) {
  int bq = blockIdx.z;
  int lane = threadIdx.x & 31, wid = threadIdx.x >> 5;
  int c0 = blockIdx.y * 64;
  int n0 = blockIdx.x * 64;
  int mt = wid & 3;
  int ntb = (wid >> 2) * 2;     // 2 n-tiles per wave
  const _Float16* Pb = P + (size_t)bq * C_ * C_;
  const _Float16* vb = vT + (size_t)bq * N_ * C_;
  v8f a0 = zero8(), a1 = zero8();
  for (int k = 0; k < C_; k += 32) {
    v16h a  = loadA(Pb, C_, c0 + mt * 16, k, lane);
    v16h b0 = loadB(vb, C_, n0 + ntb * 16, k, lane);
    v16h b1 = loadB(vb, C_, n0 + (ntb + 1) * 16, k, lane);
    a0 = wmma_f16(a, b0, a0);
    a1 = wmma_f16(a, b1, a1);
  }
  int mbase = c0 + mt * 16 + ((lane >> 4) << 3);
  int nA = n0 + ntb * 16 + (lane & 15);
  int nB = n0 + (ntb + 1) * 16 + (lane & 15);
  v8h h0, h1;
#pragma unroll
  for (int r = 0; r < 8; ++r) { h0[r] = (_Float16)a0[r]; h1[r] = (_Float16)a1[r]; }
  *(v8h*)(oT + ((size_t)bq * N_ + nA) * C_ + mbase) = h0;
  *(v8h*)(oT + ((size_t)bq * N_ + nB) * C_ + mbase) = h1;
}

// ---------------- 7) output projection -> d_out (pre-BN, final layout) -----
__global__ void k_projp(const _Float16* __restrict__ oT,
                        const _Float16* __restrict__ Whp,
                        const float* __restrict__ bip, float* __restrict__ out) {
  int bq = blockIdx.z, b = bq >> 2, q = bq & 3;
  int lane = threadIdx.x & 31, wid = threadIdx.x >> 5;
  int o0 = blockIdx.y * 32 + (wid & 1) * 16;
  int n0 = blockIdx.x * 64 + (wid >> 1) * 16;
  const _Float16* ob = oT + (size_t)bq * N_ * C_;
  v8f acc = zero8();
  for (int k = 0; k < C_; k += 32) {
    v16h a  = loadA(Whp, C_, o0, k, lane);
    v16h bf = loadB(ob, C_, n0, k, lane);
    acc = wmma_f16(a, bf, acc);
  }
  int n = n0 + (lane & 15);
  int mbase = o0 + ((lane >> 4) << 3);
#pragma unroll
  for (int r = 0; r < 8; ++r) {
    int o = mbase + r;
    out[(((size_t)b * C_ + o) * Q_ + q) * N_ + n] = acc[r] + bip[o];
  }
}

// ---------------- 8) per-channel mean/var ----------------------------------
__global__ void k_stats(const float* __restrict__ out, float* __restrict__ mean,
                        float* __restrict__ var) {
  __shared__ float s1[256], s2[256];
  int o = blockIdx.x, t = threadIdx.x;
  float sum = 0.0f, sq = 0.0f;
  const int per = B_ * Q_ * N_;  // 51200
  for (int i = t; i < per; i += 256) {
    int n = i % N_, bqi = i / N_, q = bqi & 3, b = bqi >> 2;
    float v = out[(((size_t)b * C_ + o) * Q_ + q) * N_ + n];
    sum += v;
    sq += v * v;
  }
  s1[t] = sum; s2[t] = sq;
  __syncthreads();
#pragma unroll
  for (int s = 128; s > 0; s >>= 1) {
    if (t < s) { s1[t] += s1[t + s]; s2[t] += s2[t + s]; }
    __syncthreads();
  }
  if (t == 0) {
    float m = s1[0] / per;
    mean[o] = m;
    var[o] = s2[0] / per - m * m;
  }
}

// ---------------- 9) in-place batch-norm -----------------------------------
__global__ void k_bn(float* __restrict__ out, const float* __restrict__ mean,
                     const float* __restrict__ var, const float* __restrict__ gq,
                     const float* __restrict__ beta) {
  size_t i = (size_t)blockIdx.x * blockDim.x + threadIdx.x;
  const size_t total = (size_t)B_ * C_ * Q_ * N_;
  if (i >= total) return;
  int o = (int)((i / ((size_t)Q_ * N_)) % C_);
  out[i] = (out[i] - mean[o]) * rsqrtf(var[o] + EPSV) * gq[o] + beta[o];
}

// ---------------- host-side launcher ---------------------------------------
extern "C" void kernel_launch(void* const* d_in, const int* in_sizes, int n_in,
                              void* d_out, int out_size, void* d_ws, size_t ws_size,
                              hipStream_t stream) {
  const float* x    = (const float*)d_in[0];
  const float* Wq   = (const float*)d_in[1];
  const float* bq   = (const float*)d_in[2];
  const float* Wk   = (const float*)d_in[3];
  const float* bk   = (const float*)d_in[4];
  const float* Wv   = (const float*)d_in[5];
  const float* bv   = (const float*)d_in[6];
  const float* Wp   = (const float*)d_in[7];
  const float* bp   = (const float*)d_in[8];
  const float* gam  = (const float*)d_in[9];
  const float* beta = (const float*)d_in[10];
  float* out = (float*)d_out;

  // workspace layout (bytes); total ~135.8 MB
  char* w = (char*)d_ws;
  const size_t SZ_BQNC = (size_t)BQ_ * N_ * C_ * sizeof(_Float16);  // 26,214,400
  const size_t SZ_P    = (size_t)BQ_ * C_ * C_ * sizeof(_Float16);  //  4,194,304
  const size_t SZ_W    = (size_t)C_ * C_ * sizeof(_Float16);        //    131,072
  _Float16* xh  = (_Float16*)(w);
  _Float16* qh  = (_Float16*)(w + SZ_BQNC);
  _Float16* kh  = (_Float16*)(w + 2 * SZ_BQNC);
  _Float16* vT  = (_Float16*)(w + 3 * SZ_BQNC);
  _Float16* oT  = (_Float16*)(w + 4 * SZ_BQNC);
  _Float16* P   = (_Float16*)(w + 5 * SZ_BQNC);
  _Float16* Whq = (_Float16*)(w + 5 * SZ_BQNC + SZ_P);
  _Float16* Whk = (_Float16*)((char*)Whq + SZ_W);
  _Float16* Whv = (_Float16*)((char*)Whk + SZ_W);
  _Float16* Whp = (_Float16*)((char*)Whv + SZ_W);
  float* gq_q   = (float*)((char*)Whp + SZ_W);
  float* meanb  = (float*)((char*)gq_q + 1024);
  float* varb   = (float*)((char*)meanb + 1024);

  // 1-2) quantize weights + gamma
  k_quant_w<<<dim3(C_), dim3(256), 0, stream>>>(Wq, Whq);
  k_quant_w<<<dim3(C_), dim3(256), 0, stream>>>(Wk, Whk);
  k_quant_w<<<dim3(C_), dim3(256), 0, stream>>>(Wv, Whv);
  k_quant_w<<<dim3(C_), dim3(256), 0, stream>>>(Wp, Whp);
  k_quant_gamma<<<dim3(1), dim3(256), 0, stream>>>(gam, gq_q);

  // 3) x -> f16 transposed [bq][n][c] (async global->LDS staging)
  k_xpose<<<dim3(N_ / 32, C_ / 32, BQ_), dim3(32, 8), 0, stream>>>(x, xh);

  // 4) fused q/k/v projection (WMMA)
  k_proj_qkv<<<dim3(N_ / 64, C_ / 32, BQ_), dim3(256), 0, stream>>>(
      xh, Whq, Whk, Whv, bq, bk, bv, qh, kh, vT);

  // 5) scores + softmax (WMMA + 64KB dynamic LDS)
  k_attn1<<<dim3(C_ / 64, BQ_), dim3(256), 64 * C_ * sizeof(float), stream>>>(qh, kh, P);

  // 6) P @ V (WMMA), transposed store
  k_attn2<<<dim3(N_ / 64, C_ / 64, BQ_), dim3(256), 0, stream>>>(P, vT, oT);

  // 7) output projection -> d_out pre-BN
  k_projp<<<dim3(N_ / 64, C_ / 32, BQ_), dim3(256), 0, stream>>>(oT, Whp, bp, out);

  // 8-9) batch-norm stats + in-place normalize
  k_stats<<<dim3(C_), dim3(256), 0, stream>>>(out, meanb, varb);
  const size_t total = (size_t)B_ * C_ * Q_ * N_;
  k_bn<<<dim3((unsigned)((total + 255) / 256)), dim3(256), 0, stream>>>(
      out, meanb, varb, gq_q, beta);
  (void)in_sizes; (void)n_in; (void)out_size; (void)ws_size;
}
